// SteepestDescent_7232724927248
// MI455X (gfx1250) — matrix operand order repaired and checked by reading, not compiled, not run on hardware
//
#include <hip/hip_runtime.h>
#include <hip/hip_bf16.h>

// ---- CDNA5 WMMA types (probe-confirmed signatures) ----
typedef __attribute__((ext_vector_type(16))) __bf16        v16bf;
typedef __attribute__((ext_vector_type(8)))  float         v8f;
typedef __attribute__((ext_vector_type(4)))  unsigned int  uint4v;
typedef __attribute__((ext_vector_type(4)))  unsigned short ushort4v;

union V16 { v16bf bf; uint4v q[2]; };

// f32 -> bf16 (round-to-nearest-even) without relying on __bf16 conversion codegen
__device__ __forceinline__ unsigned short f2bf_bits(float f) {
  unsigned u = __builtin_bit_cast(unsigned, f);
  u += 0x7FFFu + ((u >> 16) & 1u);
  return (unsigned short)(u >> 16);
}
__device__ __forceinline__ float bf2f(unsigned short h) {
  unsigned u = ((unsigned)h) << 16;
  return __builtin_bit_cast(float, u);
}

// Intra-wave ordering of LDS traffic (DS ops are in-order per wave; this makes
// cross-lane LDS results visible and pins compiler ordering).
__device__ __forceinline__ void lds_fence() {
  asm volatile("s_wait_dscnt 0\n\ts_wait_loadcnt 0" ::: "memory");
}

__device__ __forceinline__ float wave_sum(float v) {
#pragma unroll
  for (int m = 16; m >= 1; m >>= 1) v += __shfl_xor(v, m, 32);
  return v;
}

// 64x64 matvec in bf16x3 split precision using v_wmma_f32_16x16x32_bf16.
// A (split: Ah/Al, row-major, row stride 72 bf16) lives in LDS.
// Vector (split vh/vl, 64 bf16 each) in LDS. Result (64 f32) written to outv.
__device__ __forceinline__ void matvec64_bf16x3(
    const unsigned short* Ah, const unsigned short* Al,
    const unsigned short* vh, const unsigned short* vl,
    float* outv, int lane)
{
  const int half = lane >> 4;           // which 16-lane half of the wave
  const uint4v zq = {0u, 0u, 0u, 0u};

  // ---- B operands: column 0 holds the vector, columns 1..15 are zero.
  // B k order used here (contiguous, same as A): lane = n + 16*(k/16); VGPR r holds K = 2r,2r+1.
  V16 bh0, bl0, bh1, bl1;
  bh0.q[0] = zq; bh0.q[1] = zq; bl0.q[0] = zq; bl0.q[1] = zq;
  bh1.q[0] = zq; bh1.q[1] = zq; bl1.q[0] = zq; bl1.q[1] = zq;
  if ((lane & 15) == 0) {
    const int o0 = half * 16;           // K-chunk 0: this lane covers K o0..o0+15
    const int o1 = 32 + half * 16;      // K-chunk 1
    bh0.q[0] = *(const uint4v*)(vh + o0); bh0.q[1] = *(const uint4v*)(vh + o0 + 8);
    bl0.q[0] = *(const uint4v*)(vl + o0); bl0.q[1] = *(const uint4v*)(vl + o0 + 8);
    bh1.q[0] = *(const uint4v*)(vh + o1); bh1.q[1] = *(const uint4v*)(vh + o1 + 8);
    bl1.q[0] = *(const uint4v*)(vl + o1); bl1.q[1] = *(const uint4v*)(vl + o1 + 8);
  }

  // ---- A operand fragment offsets (contiguous k order, same as B):
  // lanes 0-15:  M=lane,     K = 0..7 (VGPR0-3) and 8..15 (VGPR4-7)
  // lanes 16-31: M=lane-16,  K = 16..23          and 24..31
  const int mloc = lane & 15;
  const int ks0 = half * 16;
  const int ks1 = half * 16 + 8;

#pragma unroll
  for (int t = 0; t < 4; ++t) {         // four 16-row output tiles
    const unsigned short* rh = Ah + (t * 16 + mloc) * 72;
    const unsigned short* rl = Al + (t * 16 + mloc) * 72;
    v8f acc = {};
#pragma unroll
    for (int kc = 0; kc < 64; kc += 32) {
      V16 ah, al;
      ah.q[0] = *(const uint4v*)(rh + kc + ks0);
      ah.q[1] = *(const uint4v*)(rh + kc + ks1);
      al.q[0] = *(const uint4v*)(rl + kc + ks0);
      al.q[1] = *(const uint4v*)(rl + kc + ks1);
      const V16& bh = kc ? bh1 : bh0;
      const V16& bl = kc ? bl1 : bl0;
      // bf16x3 split product, all chained into the f32 accumulator
      acc = __builtin_amdgcn_wmma_f32_16x16x32_bf16(false, ah.bf, false, bh.bf, (short)0, acc, false, false);
      acc = __builtin_amdgcn_wmma_f32_16x16x32_bf16(false, ah.bf, false, bl.bf, (short)0, acc, false, false);
      acc = __builtin_amdgcn_wmma_f32_16x16x32_bf16(false, al.bf, false, bh.bf, (short)0, acc, false, false);
    }
    // D column 0 lives in lanes 0 and 16: VGPR r -> row = t*16 + half*8 + r
    if ((lane & 15) == 0) {
      float* o = outv + t * 16 + half * 8;
#pragma unroll
      for (int r = 0; r < 8; ++r) o[r] = acc[r];
    }
  }
}

__device__ __forceinline__ void split_store_vec(unsigned short* vh, unsigned short* vl,
                                                int lane, float v0, float v1) {
  const unsigned short h0 = f2bf_bits(v0);
  const unsigned short l0 = f2bf_bits(v0 - bf2f(h0));
  const unsigned short h1 = f2bf_bits(v1);
  const unsigned short l1 = f2bf_bits(v1 - bf2f(h1));
  vh[lane] = h0; vh[lane + 32] = h1;
  vl[lane] = l0; vl[lane + 32] = l1;
}

#define WPB 2   // batches (waves) per 64-thread block

__global__ __launch_bounds__(64) void sd_kernel(
    const float* __restrict__ A, const float* __restrict__ bvec,
    const float* __restrict__ xin, const int* __restrict__ itp,
    float* __restrict__ out, int nbatch)
{
  // Per-wave LDS: split-bf16 copy of this batch's A (row stride padded to 72
  // bf16 = 36 dwords -> conflict-free ds_load_b128 across lanes).
  __shared__ unsigned short AhS[WPB][64 * 72];
  __shared__ unsigned short AlS[WPB][64 * 72];
  __shared__ unsigned short VhS[WPB][64];
  __shared__ unsigned short VlS[WPB][64];
  __shared__ float          OvS[WPB][64];

  const int lane = threadIdx.x & 31;
  const int wid  = threadIdx.x >> 5;
  const int batch = blockIdx.x * WPB + wid;
  if (batch >= nbatch) return;

  unsigned short* ah = AhS[wid];
  unsigned short* al = AlS[wid];
  unsigned short* vh = VhS[wid];
  unsigned short* vl = VlS[wid];
  float*          ov = OvS[wid];

  // ---- Phase 1: stream A once (b128 coalesced), split to bf16 hi/lo in LDS.
  const float* Ab = A + (size_t)batch * 4096;
#pragma unroll 4
  for (int i = 0; i < 32; ++i) {
    const int e = i * 128 + lane * 4;
    const float4 f = *(const float4*)(Ab + e);
    const int row = e >> 6, col = e & 63;
    ushort4v h, l;
    h.x = f2bf_bits(f.x); l.x = f2bf_bits(f.x - bf2f(h.x));
    h.y = f2bf_bits(f.y); l.y = f2bf_bits(f.y - bf2f(h.y));
    h.z = f2bf_bits(f.z); l.z = f2bf_bits(f.z - bf2f(h.z));
    h.w = f2bf_bits(f.w); l.w = f2bf_bits(f.w - bf2f(h.w));
    *(ushort4v*)(ah + row * 72 + col) = h;
    *(ushort4v*)(al + row * 72 + col) = l;
  }

  const int gi = batch * 64 + lane;
  const float bv0 = bvec[gi], bv1 = bvec[gi + 32];
  float x0 = xin[gi], x1 = xin[gi + 32];
  const int iters = itp[0];

  // ---- y = A x  (one WMMA matvec); thereafter y is updated by recurrence,
  // so each iteration needs exactly ONE matvec (Ag) instead of two.
  split_store_vec(vh, vl, lane, x0, x1);
  lds_fence();
  matvec64_bf16x3(ah, al, vh, vl, ov, lane);
  lds_fence();
  float y0 = ov[lane], y1 = ov[lane + 32];

  for (int it = 0; it < iters; ++it) {
    const float g0 = y0 - bv0, g1 = y1 - bv1;        // g = Ax - b
    split_store_vec(vh, vl, lane, g0, g1);
    lds_fence();
    matvec64_bf16x3(ah, al, vh, vl, ov, lane);       // Ag
    lds_fence();
    const float ag0 = ov[lane], ag1 = ov[lane + 32];
    const float num = wave_sum(g0 * g0 + g1 * g1);   // g.g
    const float den = wave_sum(g0 * ag0 + g1 * ag1); // g.Ag
    const float alpha = num / den;
    x0 -= alpha * g0;  x1 -= alpha * g1;
    y0 -= alpha * ag0; y1 -= alpha * ag1;            // y = A x (exact recurrence)
  }

  out[gi] = x0;
  out[gi + 32] = x1;
}

extern "C" void kernel_launch(void* const* d_in, const int* in_sizes, int n_in,
                              void* d_out, int out_size, void* d_ws, size_t ws_size,
                              hipStream_t stream) {
  const float* A  = (const float*)d_in[0];
  const float* b  = (const float*)d_in[1];
  const float* x  = (const float*)d_in[2];
  const int*   it = (const int*)d_in[3];
  float* out = (float*)d_out;
  const int nbatch = in_sizes[1] / 64;               // b is [B, 64]
  const int blocks = (nbatch + WPB - 1) / WPB;
  sd_kernel<<<blocks, 64, 0, stream>>>(A, b, x, it, out, nbatch);
}
